// AnalysisBand_34488587387388
// MI455X (gfx1250) — matrix operand order repaired and checked by reading, not compiled, not run on hardware
//
#include <hip/hip_runtime.h>
#include <math.h>
#include <stdint.h>

#define N_T   32768
#define NCH   128
#define NATOM 512
#define ATOM  512
#define OUTLEN (N_T + 1023)

typedef __attribute__((ext_vector_type(16))) _Float16 v16h;
typedef __attribute__((ext_vector_type(8)))  _Float16 v8h;
typedef __attribute__((ext_vector_type(8)))  float    v8f;
typedef _Float16 h16;

// ---------------- WMMA helpers (CDNA5 wave32, 16x16x32 f16 -> f32) ----------

__device__ inline v8f wmma16(v16h a, v16h b, v8f c) {
  return __builtin_amdgcn_wmma_f32_16x16x32_f16(false, a, false, b, (short)0, c,
                                                false, false);
}

// A fragment: 16x32 f16, row m = lane&15; per-lane one contiguous 16-f16 run at k0 + 16*half.
__device__ inline v16h loadA(const h16* W, int ldw, int m, int half, int k0) {
  const h16* r = W + (size_t)m * ldw + k0 + half * 16;
  v8h lo = *(const v8h*)r;
  v8h hi = *(const v8h*)(r + 8);
  v16h a;
#pragma unroll
  for (int e = 0; e < 8; ++e) { a[e] = lo[e]; a[8 + e] = hi[e]; }
  return a;
}

// B fragment: per-lane contiguous 16-f16 run from LDS [t][ci] tile.
__device__ inline v16h loadB(const h16* p) {
  v16h b;
  __builtin_memcpy(&b, p, sizeof(b));
  return b;
}

// ---------------- CDNA5 async memory->LDS copy (ASYNCcnt tracked) -----------
// Each active lane copies 16 contiguous bytes from global to its LDS address.
// LDS generic address low 32 bits == LDS byte offset (flat aperture rule).

__device__ inline unsigned lds_off(const void* p) {
  return (unsigned)(uintptr_t)p;
}
__device__ inline void async_b128(unsigned lds_byte, const h16* g) {
  asm volatile("global_load_async_to_lds_b128 %0, %1, off"
               :: "v"(lds_byte), "v"((unsigned long long)(uintptr_t)g)
               : "memory");
}
__device__ inline void async_wait0() {
  asm volatile("s_wait_asynccnt 0" ::: "memory");
}

// ---------------- Weight prep (f32 -> f16, relayout) ------------------------

__global__ void k_prep(const float* bank, const float* reduce_w, const float* dil,
                       const float* cw, const float* taw, h16* bankW, h16* Wm,
                       h16* Wp, h16* Wd, h16* Wc, h16* Wa) {
  int i = blockIdx.x * blockDim.x + threadIdx.x;
  if (i < 65536) { bankW[i] = (h16)bank[i]; return; }              // [128][512]
  i -= 65536;
  if (i < 16384) {                                                 // Wm [128][128]
    int co = i >> 7, ci = i & 127;
    Wm[i] = (h16)reduce_w[co * 161 + ci];
    return;
  }
  i -= 16384;
  if (i < 8192) {                                                  // Wp [128][64] zero-padded
    int co = i >> 6, j = i & 63;
    Wp[i] = (h16)((j < 33) ? reduce_w[co * 161 + 128 + j] : 0.f);
    return;
  }
  i -= 8192;
  if (i < 294912) {                                                // Wd [6][3][128][128]
    int ci = i & 127, co = (i >> 7) & 127, q = i >> 14;
    int kk = q % 3, ii = q / 3;
    Wd[i] = (h16)dil[(((ii * 128 + co) * 128 + ci) * 3) + kk];
    return;
  }
  i -= 294912;
  if (i < 98304) { Wc[i] = (h16)cw[i]; return; }                   // [6][128][128]
  i -= 98304;
  if (i < 65536) { Wa[i] = (h16)taw[i]; return; }                  // [512][128]
}

__global__ void k_rowsum(const float* reduce_w, float* rowsum) {
  int co = threadIdx.x;
  if (co < 128) {
    float s = 0.f;
    for (int ci = 0; ci < 128; ++ci) s += reduce_w[co * 161 + ci];
    rowsum[co] = s;
  }
}

__global__ void k_zero(float* out, long n, float* stats, int ns) {
  long i = (long)blockIdx.x * blockDim.x + threadIdx.x;
  if (i < n) out[i] = 0.f;
  if (i < ns) stats[i] = 0.f;
}

__global__ void k_x16(const float* x, h16* x16, int n) {
  int i = blockIdx.x * blockDim.x + threadIdx.x;
  if (i < n) x16[i] = (h16)x[i];
}

// ---------------- Bank conv: 1 -> 128 channels, causal k=512 ----------------
// H0[b][t][co] = sum_k bank[co,k] * x[t + k - 512]

__global__ void k_bank(const h16* x16, const h16* bankW, h16* H0) {
  __shared__ h16 xs[576];  // x[t0-512 .. t0+63]
  int tid = threadIdx.x, wave = tid >> 5, lane = tid & 31;
  int half = lane >> 4, col = lane & 15;
  int b = blockIdx.y, t0 = blockIdx.x * 64;
  // stage window: 72 x 16B async copies; causal prefix zero-filled
  if (tid < 72) {
    int g0 = t0 - 512 + tid * 8;
    if (g0 >= 0) {
      async_b128(lds_off(xs) + tid * 16, x16 + (size_t)b * N_T + g0);
    } else {
      v8h z8 = {};
      *(v8h*)&xs[tid * 8] = z8;
    }
  }
  async_wait0();
  __syncthreads();
  int m = wave * 16 + col;
  v8f z = {0.f, 0.f, 0.f, 0.f, 0.f, 0.f, 0.f, 0.f};
  v8f acc[4] = {z, z, z, z};
  for (int kc = 0; kc < 16; ++kc) {
    v16h a = loadA(bankW, 512, m, half, kc * 32);
#pragma unroll
    for (int tt = 0; tt < 4; ++tt) {
      v16h bm = loadB(&xs[tt * 16 + col + kc * 32 + half * 16]);
      acc[tt] = wmma16(a, bm, acc[tt]);
    }
  }
#pragma unroll
  for (int tt = 0; tt < 4; ++tt) {
    v8h o;
#pragma unroll
    for (int r = 0; r < 8; ++r) o[r] = (h16)acc[tt][r];
    *(v8h*)&H0[((size_t)b * N_T + t0 + tt * 16 + col) * NCH + wave * 16 +
               half * 8] = o;
  }
}

// ---------------- Per-batch mean/var over (C,T) -----------------------------

__global__ void k_stats(const h16* H0, float* stats) {
  __shared__ float ss[256], sq[256];
  int b = blockIdx.y, tid = threadIdx.x;
  size_t base = (size_t)b * NCH * N_T + (size_t)blockIdx.x * 8192;
  float s = 0.f, q = 0.f;
  for (int i = tid; i < 8192; i += 256) {
    float v = (float)H0[base + i];
    s += v;
    q += v * v;
  }
  ss[tid] = s; sq[tid] = q;
  __syncthreads();
  for (int st = 128; st > 0; st >>= 1) {
    if (tid < st) { ss[tid] += ss[tid + st]; sq[tid] += sq[tid + st]; }
    __syncthreads();
  }
  if (tid == 0) {
    atomicAdd(&stats[b * 2 + 0], ss[0]);
    atomicAdd(&stats[b * 2 + 1], sq[0]);
  }
}

// ---------------- Positional features: P[t][64] (33 rows used) --------------

__global__ void k_pos(h16* P) {
  int t = blockIdx.x * blockDim.x + threadIdx.x;
  if (t >= N_T) return;
  float p = -1.f + 2.f * (float)t / (float)(N_T - 1);
  h16* row = P + (size_t)t * 64;
  row[0] = (h16)p;
  for (int i = 0; i < 16; ++i) {
    float s = p * exp2f((float)i);
    row[1 + 2 * i] = (h16)sinf(s);
    row[2 + 2 * i] = (h16)cosf(s);
  }
  for (int r = 33; r < 64; ++r) row[r] = (h16)0.f;
}

// ---------------- Reduce conv (161->128), normalization folded --------------
// y = invstd * (Wm @ h) + (Wp @ pos) - mean*invstd*rowsum(Wm)

__global__ void k_reduce(const h16* H0, const h16* P, const h16* Wm,
                         const h16* Wp, const float* rowsum, const float* stats,
                         h16* HA) {
  __shared__ h16 xt[64 * 128];  // [t][ci]
  __shared__ h16 pt[64 * 64];
  int tid = threadIdx.x, wave = tid >> 5, lane = tid & 31;
  int half = lane >> 4, col = lane & 15;
  int b = blockIdx.y, t0 = blockIdx.x * 64;
  // async stage: 64 rows x 256B (xt), 64 rows x 128B (pt)
  for (int i = tid; i < 64 * 16; i += 256) {
    int lt = i >> 4, seg = i & 15;
    async_b128(lds_off(xt) + lt * 256 + seg * 16,
               H0 + ((size_t)b * N_T + t0 + lt) * NCH + seg * 8);
  }
  for (int i = tid; i < 64 * 8; i += 256) {
    int lt = i >> 3, seg = i & 7;
    async_b128(lds_off(pt) + lt * 128 + seg * 16,
               P + (size_t)(t0 + lt) * 64 + seg * 8);
  }
  async_wait0();
  __syncthreads();
  float M = (float)NCH * (float)N_T;
  float mean = stats[b * 2 + 0] / M;
  float var = fmaxf(stats[b * 2 + 1] / M - mean * mean, 0.f);
  float invstd = 1.f / (sqrtf(var) + 1e-8f);
  int m = wave * 16 + col;
  v8f z = {0.f, 0.f, 0.f, 0.f, 0.f, 0.f, 0.f, 0.f};
  v8f am[4] = {z, z, z, z}, ap[4] = {z, z, z, z};
  for (int c0 = 0; c0 < 128; c0 += 32) {
    v16h a = loadA(Wm, 128, m, half, c0);
#pragma unroll
    for (int tt = 0; tt < 4; ++tt) {
      v16h bm = loadB(&xt[(tt * 16 + col) * 128 + c0 + half * 16]);
      am[tt] = wmma16(a, bm, am[tt]);
    }
  }
  for (int c0 = 0; c0 < 64; c0 += 32) {
    v16h a = loadA(Wp, 64, m, half, c0);
#pragma unroll
    for (int tt = 0; tt < 4; ++tt) {
      v16h bm = loadB(&pt[(tt * 16 + col) * 64 + c0 + half * 16]);
      ap[tt] = wmma16(a, bm, ap[tt]);
    }
  }
#pragma unroll
  for (int tt = 0; tt < 4; ++tt) {
    v8h o;
#pragma unroll
    for (int r = 0; r < 8; ++r) {
      int row = wave * 16 + half * 8 + r;
      float v = am[tt][r] * invstd + ap[tt][r] - mean * invstd * rowsum[row];
      o[r] = (h16)v;
    }
    *(v8h*)&HA[((size_t)b * N_T + t0 + tt * 16 + col) * NCH + wave * 16 +
               half * 8] = o;
  }
}

// ---------------- Dilated conv k=3 (3 shifted GEMMs accumulated) ------------
// U[b][t][co] = sum_k sum_ci Wd[k][co][ci] * h[ci, t + (k-1)*d]

__global__ void k_dilated(const h16* Hin, const h16* Wdk, h16* U, int d) {
  __shared__ h16 xt[226 * 32];  // [t0-d .. t0+63+d][32-ci chunk]
  int tid = threadIdx.x, wave = tid >> 5, lane = tid & 31;
  int half = lane >> 4, col = lane & 15;
  int b = blockIdx.y, t0 = blockIdx.x * 64;
  int Wt = 64 + 2 * d;
  int m = wave * 16 + col;
  v8f z = {0.f, 0.f, 0.f, 0.f, 0.f, 0.f, 0.f, 0.f};
  v8f acc[4] = {z, z, z, z};
  for (int c0 = 0; c0 < 128; c0 += 32) {
    __syncthreads();
    // async stage: Wt rows x 64B (4 x 16B segments), halo zero-filled
    for (int i = tid; i < Wt * 4; i += 256) {
      int lt = i >> 2, seg = i & 3;
      int g = t0 - d + lt;
      if (g >= 0 && g < N_T) {
        async_b128(lds_off(xt) + lt * 64 + seg * 16,
                   Hin + ((size_t)b * N_T + g) * NCH + c0 + seg * 8);
      } else {
        v8h z8 = {};
        *(v8h*)&xt[lt * 32 + seg * 8] = z8;
      }
    }
    async_wait0();
    __syncthreads();
    for (int k = 0; k < 3; ++k) {
      v16h a = loadA(Wdk + (size_t)k * NCH * NCH, 128, m, half, c0);
#pragma unroll
      for (int tt = 0; tt < 4; ++tt) {
        v16h bm = loadB(&xt[(tt * 16 + col + k * d) * 32 + half * 16]);
        acc[tt] = wmma16(a, bm, acc[tt]);
      }
    }
  }
#pragma unroll
  for (int tt = 0; tt < 4; ++tt) {
    v8h o;
#pragma unroll
    for (int r = 0; r < 8; ++r) o[r] = (h16)acc[tt][r];
    *(v8h*)&U[((size_t)b * N_T + t0 + tt * 16 + col) * NCH + wave * 16 +
              half * 8] = o;
  }
}

// ---------------- 1x1 conv + residual + leaky ReLU (in-place on HA) ---------

__global__ void k_pw(const h16* U, const h16* Wc1, h16* HA) {
  __shared__ h16 ut[64 * 128];
  int tid = threadIdx.x, wave = tid >> 5, lane = tid & 31;
  int half = lane >> 4, col = lane & 15;
  int b = blockIdx.y, t0 = blockIdx.x * 64;
  for (int i = tid; i < 64 * 16; i += 256) {
    int lt = i >> 4, seg = i & 15;
    async_b128(lds_off(ut) + lt * 256 + seg * 16,
               U + ((size_t)b * N_T + t0 + lt) * NCH + seg * 8);
  }
  async_wait0();
  __syncthreads();
  int m = wave * 16 + col;
  v8f z = {0.f, 0.f, 0.f, 0.f, 0.f, 0.f, 0.f, 0.f};
  v8f acc[4] = {z, z, z, z};
  for (int c0 = 0; c0 < 128; c0 += 32) {
    v16h a = loadA(Wc1, 128, m, half, c0);
#pragma unroll
    for (int tt = 0; tt < 4; ++tt) {
      v16h bm = loadB(&ut[(tt * 16 + col) * 128 + c0 + half * 16]);
      acc[tt] = wmma16(a, bm, acc[tt]);
    }
  }
#pragma unroll
  for (int tt = 0; tt < 4; ++tt) {
    size_t o = ((size_t)b * N_T + t0 + tt * 16 + col) * NCH + wave * 16 +
               half * 8;
    v8h hv = *(v8h*)&HA[o];
    v8h ov;
#pragma unroll
    for (int r = 0; r < 8; ++r) {
      float v = acc[tt][r] + (float)hv[r];
      v = (v >= 0.f) ? v : 0.2f * v;
      ov[r] = (h16)v;
    }
    *(v8h*)&HA[o] = ov;
  }
}

// ---------------- context = max over time (coalesced two-stage) -------------

__global__ void k_ctx1(const h16* HA, float* part) {
  __shared__ float pm[256];
  int b = blockIdx.y, chunk = blockIdx.x, tid = threadIdx.x;
  size_t base = ((size_t)b * N_T + (size_t)chunk * 4096) * NCH;
  float m = -3.4e38f;
  // thread's channel is fixed: ci = tid & 127 (256 % 128 == 0)
  for (int i = tid; i < 4096 * NCH; i += 256) m = fmaxf(m, (float)HA[base + i]);
  pm[tid] = m;
  __syncthreads();
  if (tid < 128)
    part[((size_t)b * 8 + chunk) * NCH + tid] = fmaxf(pm[tid], pm[tid + 128]);
}

__global__ void k_ctx2(const float* part, float* ctx, int total) {
  int i = blockIdx.x * blockDim.x + threadIdx.x;
  if (i >= total) return;
  int b = i >> 7, c = i & 127;
  float m = -3.4e38f;
  for (int ch = 0; ch < 8; ++ch)
    m = fmaxf(m, part[((size_t)b * 8 + ch) * NCH + c]);
  ctx[i] = m;
}

// ---------------- to_atoms (128->512) + ReLU + block-local top-64 -----------

__global__ void k_atoms(const h16* HA, const h16* Wa, float* candV, int* candI) {
  __shared__ h16 ft[16 * 128];
  __shared__ float buf[8192];  // [ao=512][t=16]
  __shared__ float rmax[256];
  __shared__ int ridx[256];
  int tid = threadIdx.x, wave = tid >> 5, lane = tid & 31;
  int half = lane >> 4, col = lane & 15;
  int b = blockIdx.y, t0 = blockIdx.x * 16;
  {
    int lt = tid >> 4, seg = tid & 15;  // exactly 256 segments
    async_b128(lds_off(ft) + lt * 256 + seg * 16,
               HA + ((size_t)b * N_T + t0 + lt) * NCH + seg * 8);
  }
  async_wait0();
  __syncthreads();
  v8f z = {0.f, 0.f, 0.f, 0.f, 0.f, 0.f, 0.f, 0.f};
  for (int j = 0; j < 4; ++j) {
    int q = wave + 8 * j;
    int m = q * 16 + col;
    v8f acc = z;
    for (int c0 = 0; c0 < 128; c0 += 32) {
      v16h a = loadA(Wa, 128, m, half, c0);
      v16h bm = loadB(&ft[col * 128 + c0 + half * 16]);
      acc = wmma16(a, bm, acc);
    }
#pragma unroll
    for (int r = 0; r < 8; ++r) {
      int ao = q * 16 + half * 8 + r;
      buf[ao * 16 + col] = fmaxf(acc[r], 0.f);
    }
  }
  __syncthreads();
  for (int it = 0; it < 64; ++it) {
    float bmv = -1.f;
    int bi = 0;
    for (int i = tid; i < 8192; i += 256)
      if (buf[i] > bmv) { bmv = buf[i]; bi = i; }
    rmax[tid] = bmv; ridx[tid] = bi;
    __syncthreads();
    for (int s = 128; s > 0; s >>= 1) {
      if (tid < s && rmax[tid + s] > rmax[tid]) {
        rmax[tid] = rmax[tid + s];
        ridx[tid] = ridx[tid + s];
      }
      __syncthreads();
    }
    if (tid == 0) {
      int wI = ridx[0];
      float v = rmax[0];
      size_t slot = ((size_t)b * (N_T / 16) + blockIdx.x) * 64 + it;
      candV[slot] = (v > 0.f) ? v : 0.f;
      candI[slot] = (wI >> 4) * N_T + t0 + (wI & 15);
      buf[wI] = -2.f;
    }
    __syncthreads();
  }
}

// ---------------- global top-64 merge per batch -----------------------------

__global__ void k_merge(float* candV, const int* candI, float* spV, int* spI) {
  __shared__ float rmax[256];
  __shared__ int ridx[256];
  int b = blockIdx.x, tid = threadIdx.x;
  const int M = (N_T / 16) * 64;
  float* cv = candV + (size_t)b * M;
  const int* ci = candI + (size_t)b * M;
  for (int it = 0; it < 64; ++it) {
    float bmv = -1.f;
    int bi = 0;
    for (int i = tid; i < M; i += 256) {
      float v = cv[i];
      if (v > bmv) { bmv = v; bi = i; }
    }
    rmax[tid] = bmv; ridx[tid] = bi;
    __syncthreads();
    for (int s = 128; s > 0; s >>= 1) {
      if (tid < s && rmax[tid + s] > rmax[tid]) {
        rmax[tid] = rmax[tid + s];
        ridx[tid] = ridx[tid + s];
      }
      __syncthreads();
    }
    if (tid == 0) {
      spV[b * 64 + it] = rmax[0];
      spI[b * 64 + it] = ci[ridx[0]];
      cv[ridx[0]] = -2.f;
    }
    __syncthreads();
  }
}

// ---------------- sparse atom synthesis (transposed-conv scatter) -----------

__global__ void k_scatter(const float* spV, const int* spI, const float* atoms,
                          float* out) {
  int b = blockIdx.y, e = blockIdx.x, tid = threadIdx.x;
  float v = spV[b * 64 + e];
  if (v <= 0.f) return;
  int idx = spI[b * 64 + e];
  int ao = idx / N_T;
  int t = idx - ao * N_T;
  float* dst = out + (size_t)b * OUTLEN + t + 512;
  const float* ar = atoms + (size_t)ao * ATOM;
  for (int j = tid; j < ATOM; j += 256) atomicAdd(&dst[j], v * ar[j]);
}

// ---------------- host launch ----------------------------------------------

extern "C" void kernel_launch(void* const* d_in, const int* in_sizes, int n_in,
                              void* d_out, int out_size, void* d_ws,
                              size_t ws_size, hipStream_t stream) {
  const float* x = (const float*)d_in[0];
  const float* bank = (const float*)d_in[1];
  const float* atoms = (const float*)d_in[2];
  const float* reduce_w = (const float*)d_in[3];
  const float* taw = (const float*)d_in[4];
  const float* dil = (const float*)d_in[5];
  const float* cwts = (const float*)d_in[6];
  int B = in_sizes[0] / N_T;
  (void)n_in; (void)ws_size;

  char* base = (char*)d_ws;
  size_t off = 0;
  auto alloc = [&](size_t nbytes) -> void* {
    void* p = base + off;
    off = (off + nbytes + 255) & ~(size_t)255;
    return p;
  };
  h16* H0 = (h16*)alloc((size_t)B * NCH * N_T * 2);   // [b][t][ci]
  h16* P = (h16*)alloc((size_t)N_T * 64 * 2);         // [t][64]
  h16* HA = (h16*)alloc((size_t)B * NCH * N_T * 2);
  h16* HB = (h16*)alloc((size_t)B * NCH * N_T * 2);
  h16* x16 = (h16*)alloc((size_t)B * N_T * 2);
  h16* bankW = (h16*)alloc(65536 * 2);
  h16* Wm = (h16*)alloc(16384 * 2);
  h16* Wp = (h16*)alloc(8192 * 2);
  h16* Wd = (h16*)alloc(294912 * 2);
  h16* Wc = (h16*)alloc(98304 * 2);
  h16* Wa = (h16*)alloc(65536 * 2);
  float* rowsum = (float*)alloc(128 * 4);
  float* stats = (float*)alloc(64);
  float* part = (float*)alloc((size_t)B * 8 * NCH * 4);
  float* candV = (float*)alloc((size_t)B * (N_T / 16) * 64 * 4);
  int* candI = (int*)alloc((size_t)B * (N_T / 16) * 64 * 4);
  float* spV = (float*)alloc((size_t)B * 64 * 4);
  int* spI = (int*)alloc((size_t)B * 64 * 4);

  float* out = (float*)d_out;
  float* ctx = out + (size_t)B * OUTLEN;

  {
    long n = (long)out_size;
    int g = (int)((n + 255) / 256);
    k_zero<<<g, 256, 0, stream>>>(out, n, stats, 2 * B);
  }
  k_prep<<<(548864 + 255) / 256, 256, 0, stream>>>(bank, reduce_w, dil, cwts,
                                                   taw, bankW, Wm, Wp, Wd, Wc, Wa);
  k_rowsum<<<1, 128, 0, stream>>>(reduce_w, rowsum);
  k_x16<<<(B * N_T + 255) / 256, 256, 0, stream>>>(x, x16, B * N_T);
  k_bank<<<dim3(N_T / 64, B), 256, 0, stream>>>(x16, bankW, H0);
  k_stats<<<dim3(512, B), 256, 0, stream>>>(H0, stats);
  k_pos<<<N_T / 256, 256, 0, stream>>>(P);
  k_reduce<<<dim3(N_T / 64, B), 256, 0, stream>>>(H0, P, Wm, Wp, rowsum, stats,
                                                  HA);
  const int DIL[6] = {1, 3, 9, 27, 81, 1};
  for (int i = 0; i < 6; ++i) {
    k_dilated<<<dim3(N_T / 64, B), 256, 0, stream>>>(
        HA, Wd + (size_t)i * 3 * NCH * NCH, HB, DIL[i]);
    k_pw<<<dim3(N_T / 64, B), 256, 0, stream>>>(HB, Wc + (size_t)i * NCH * NCH,
                                                HA);
  }
  k_ctx1<<<dim3(8, B), 256, 0, stream>>>(HA, part);
  k_ctx2<<<(B * NCH + 255) / 256, 256, 0, stream>>>(part, ctx, B * NCH);
  k_atoms<<<dim3(N_T / 16, B), 256, 0, stream>>>(HA, Wa, candV, candI);
  k_merge<<<B, 256, 0, stream>>>(candV, candI, spV, spI);
  k_scatter<<<dim3(64, B), 256, 0, stream>>>(spV, spI, atoms, out);
}
